// MultiHeadAttention_47459388621210
// MI455X (gfx1250) — matrix operand order repaired and checked by reading, not compiled, not run on hardware
//
#include <hip/hip_runtime.h>
#include <hip/hip_bf16.h>

typedef __attribute__((ext_vector_type(16))) _Float16 v16h;
typedef __attribute__((ext_vector_type(8)))  _Float16 v8h;
typedef __attribute__((ext_vector_type(8)))  float    v8f;
typedef __attribute__((ext_vector_type(4)))  float    v4f;
typedef __attribute__((ext_vector_type(4)))  unsigned int v4ui;
typedef __attribute__((ext_vector_type(8)))  int      v8si;
typedef __attribute__((ext_vector_type(4)))  int      v4si;

#define EMB_   1024
#define HEADS_ 16
#define HD_    64
#define BATCH_ 4
#define SEQ_   2048
#define ROWS_  (BATCH_ * SEQ_)
#define SCALE_ 0.03125f  /* 1/sqrt(EMB) */

#define LDS_STRIDE_ 40                       /* 32 f16 + 8 f16 pad per row */
#define AS_BYTES_   (128 * LDS_STRIDE_ * 2)  /* one A/B buffer: 10240 B */

union V16U { v16h v; v8h h[2]; };

static __device__ __forceinline__ v8f wmma_f16(v16h a, v16h b, v8f c) {
  // D = A(16x32 f16) * B(32x16 f16) + C(16x16 f32)
  return __builtin_amdgcn_wmma_f32_16x16x32_f16(
      /*neg_a=*/false, a, /*neg_b=*/false, b,
      /*c_mod=*/(short)0, c, /*reuse_a=*/false, /*reuse_b=*/false);
}

// ---- DPP16 row_xmask butterfly (pure VALU, no LDS): exchange lane^MASK
// ---- within each 16-lane DPP row. gfx1250 DPP ctrl: 0x160 | mask.
template<int CTRL>
static __device__ __forceinline__ float dpp_mov(float x) {
  int i = __float_as_int(x);
  return __int_as_float(__builtin_amdgcn_update_dpp(i, i, CTRL, 0xF, 0xF, false));
}
static __device__ __forceinline__ float rowmax16(float v) {
  v = fmaxf(v, dpp_mov<0x161>(v));
  v = fmaxf(v, dpp_mov<0x162>(v));
  v = fmaxf(v, dpp_mov<0x164>(v));
  v = fmaxf(v, dpp_mov<0x168>(v));
  return v;
}
static __device__ __forceinline__ float rowsum16(float v) {
  v += dpp_mov<0x161>(v);
  v += dpp_mov<0x162>(v);
  v += dpp_mov<0x164>(v);
  v += dpp_mov<0x168>(v);
  return v;
}

// ---------------------------------------------------------------------------
// Tensor Data Mover: DMA a [128 rows x 32 f16] tile (row stride EMB_ f16)
// from global memory into LDS at byte offset `lds_off`, padding the LDS
// destination by 4 DWORDs after every 16 DWORDs (row stride 32 -> 40 f16).
// D# per CDNA5 ISA (group0 128b, group1 256b, 2D tensor).
// This toolchain exposes the 6-arg builtin form:
//   (uint32x4 g0, int32x8 g1, int32x4, int32x4, int32x8, i32 cpol)
// ---------------------------------------------------------------------------
static __device__ __forceinline__ void tdm_load_tile_a(const _Float16* gsrc,
                                                       unsigned lds_off) {
  unsigned long long ga = (unsigned long long)(uintptr_t)gsrc;
  v4ui g0;
  g0[0] = 1u;                                   // count=1, user desc, no gather
  g0[1] = lds_off;                              // lds_addr (bytes)
  g0[2] = (unsigned)ga;                         // global_addr[31:0]
  g0[3] = (unsigned)((ga >> 32) & 0x01FFFFFFu)  // global_addr[56:32]
        | (2u << 30);                           // type=2 ("image")
  v8si g1;
  g1[0] = (int)((1u << 16)      // data_size = 1 -> 2 bytes
              | (1u << 20)      // pad_enable
              | (3u << 22)      // pad_interval: 16 DWORDs
              | (3u << 25));    // pad_amount:   4 DWORDs
  g1[1] = (int)((EMB_ & 0xFFFF) << 16);            // tensor_dim0 lo16 @bits63:48
  g1[2] = (int)((EMB_ >> 16) & 0xFFFF);            // tensor_dim0 hi16, dim1 lo16=0
  g1[3] = (int)(1u | (32u << 16));                 // tensor_dim1=65536, tile_dim0=32
  g1[4] = 128;                                     // tile_dim1=128, tile_dim2=0
  g1[5] = EMB_;                                    // tensor_dim0_stride lo32
  g1[6] = 0;                                       // stride hi, dim1_stride lo
  g1[7] = 0;
  v4si z4 = {0, 0, 0, 0};                          // groups 2/3 unused (2D)
  v8si z8 = {0, 0, 0, 0, 0, 0, 0, 0};
  __builtin_amdgcn_tensor_load_to_lds(g0, g1, z4, z4, z8, /*cpol=*/0);
}

// ---------------------------------------------------------------------------
// f32 -> f16 bulk convert (8 elements / thread, sizes are multiples of 2048)
// ---------------------------------------------------------------------------
__global__ void __launch_bounds__(256)
cvt_f32_f16(const float* __restrict__ src, _Float16* __restrict__ dst, int n) {
  int i = (blockIdx.x * 256 + threadIdx.x) * 8;
  if (i + 8 <= n) {
    v4f a = *(const v4f*)&src[i];
    v4f b = *(const v4f*)&src[i + 4];
    v8h o;
    o[0] = (_Float16)a[0]; o[1] = (_Float16)a[1];
    o[2] = (_Float16)a[2]; o[3] = (_Float16)a[3];
    o[4] = (_Float16)b[0]; o[5] = (_Float16)b[1];
    o[6] = (_Float16)b[2]; o[7] = (_Float16)b[3];
    *(v8h*)&dst[i] = o;
  }
}

// ---------------------------------------------------------------------------
// 128x128 macro-tile GEMM:  Y[r,c] = sum_k A[r,k] * W[k,c] + bias[c]
//   8 waves, each wave owns a 64x32 tile (4x2 WMMA 16x16 tiles).
//   A tile: double-buffered TDM (tensor_load_to_lds) with HW LDS padding.
//   B tile: register double-buffered manual transpose into LDS.
//   Dynamic LDS layout (byte offsets): As[2] @ 0, Bs[2] @ 2*AS_BYTES_.
//   MODE 0: f16 out, head-split   [b][h][n][d]   (Q, K)
//   MODE 2: f16 out, head-split-T [b][h][d][n]   (V, pre-transposed for PV)
//   MODE 1: f32 out, row-major    [r][c]         (final projection)
// ---------------------------------------------------------------------------
#define AS_(buf, r, c) AsBase[((buf) * 128 + (r)) * LDS_STRIDE_ + (c)]
#define BS_(buf, r, c) BsBase[((buf) * 128 + (r)) * LDS_STRIDE_ + (c)]

template<int MODE>
__global__ void __launch_bounds__(256)
gemm128(const _Float16* __restrict__ A, const _Float16* __restrict__ W,
        const float* __restrict__ bias, void* __restrict__ outp) {
  extern __shared__ _Float16 smem[];           // 4 * AS_BYTES_ bytes
  _Float16* AsBase = smem;                      // [2][128][40]
  _Float16* BsBase = smem + 2 * 128 * LDS_STRIDE_;

  const int t       = threadIdx.x;
  const int lane    = t & 31;
  const int wave    = t >> 5;
  const int rowBase = blockIdx.x * 128;
  const int colBase = blockIdx.y * 128;
  const int wm      = (wave >> 2) * 64;  // wave M offset: 0 / 64
  const int wn      = (wave & 3) * 32;   // wave N offset: 0..96
  const int lrow    = lane & 15;
  const int khalf   = (lane >> 4) * 8;   // A-fragment K half
  const int kb2     = (lane >> 4) * 16;  // contiguous K base (unused; B fragments use khalf)

  v8f acc[4][2] = {};

  const int bk = t >> 3;          // B staging: k 0..31
  const int bn = (t & 7) * 16;    //            n 0..112

  // ---- prologue: kick TDM for A tile 0, preload B tile 0 into registers ----
  if (t < 32)
    tdm_load_tile_a(&A[(size_t)rowBase * EMB_], 0u);
  v8h b0, b1;
  {
    const v8h* gb0 = (const v8h*)&W[(size_t)bk * EMB_ + colBase + bn];
    b0 = gb0[0]; b1 = gb0[1];
  }

  for (int i = 0; i < EMB_ / 32; ++i) {
    const int kb  = i * 32;
    const int buf = i & 1;

    // ---- stage B (transposed) into Bs[buf] ----
#pragma unroll
    for (int j = 0; j < 8; ++j) {
      BS_(buf, bn + j, bk)     = b0[j];
      BS_(buf, bn + 8 + j, bk) = b1[j];
    }
    __builtin_amdgcn_s_wait_tensorcnt(0);  // As[buf] DMA done (no-op for waves 1-7)
    __syncthreads();

    // ---- issue next A-tile TDM + prefetch next B tile into registers ----
    if (kb + 32 < EMB_) {
      if (t < 32)
        tdm_load_tile_a(&A[(size_t)rowBase * EMB_ + kb + 32],
                        (unsigned)((buf ^ 1) * AS_BYTES_));
      const v8h* gbn = (const v8h*)&W[(size_t)(kb + 32 + bk) * EMB_ + colBase + bn];
      b0 = gbn[0]; b1 = gbn[1];
    }

    // ---- fragments ----
    v16h afr[4], bfr[2];
#pragma unroll
    for (int m = 0; m < 4; ++m) {
      V16U u;
      u.h[0] = *(const v8h*)&AS_(buf, wm + m * 16 + lrow, khalf);
      u.h[1] = *(const v8h*)&AS_(buf, wm + m * 16 + lrow, khalf + 16);
      afr[m] = u.v;
    }
#pragma unroll
    for (int n = 0; n < 2; ++n) {
      V16U u;
      u.h[0] = *(const v8h*)&BS_(buf, wn + n * 16 + lrow, khalf);
      u.h[1] = *(const v8h*)&BS_(buf, wn + n * 16 + lrow, khalf + 16);
      bfr[n] = u.v;
    }
#pragma unroll
    for (int m = 0; m < 4; ++m)
#pragma unroll
      for (int n = 0; n < 2; ++n)
        acc[m][n] = wmma_f16(afr[m], bfr[n], acc[m][n]);
    __syncthreads();
  }

  // ---- epilogue: C layout M = j + 8*(lane>>4), N = lane&15 ----
  const int moff = (lane >> 4) * 8;
#pragma unroll
  for (int m = 0; m < 4; ++m)
#pragma unroll
    for (int n = 0; n < 2; ++n)
#pragma unroll
      for (int j = 0; j < 8; ++j) {
        int gr = rowBase + wm + m * 16 + moff + j;
        int gc = colBase + wn + n * 16 + lrow;
        float v = acc[m][n][j] + bias[gc];
        if (MODE == 1) {
          ((float*)outp)[(size_t)gr * EMB_ + gc] = v;
        } else {
          int bi = gr / SEQ_, nn = gr % SEQ_;
          int hh = gc / HD_,  dd = gc % HD_;
          size_t idx = (MODE == 0)
              ? ((((size_t)bi * HEADS_ + hh) * SEQ_ + nn) * HD_ + dd)
              : ((((size_t)bi * HEADS_ + hh) * HD_ + dd) * SEQ_ + nn);
          ((_Float16*)outp)[idx] = (_Float16)v;
        }
      }
}

// ---------------------------------------------------------------------------
// Flash attention: one block = 64 query rows of one (b,h); 4 waves x 16 rows.
//   Q: [b][h][n][d] f16, K: [b][h][n][d] f16, Vt: [b][h][d][n] f16
//   Ah out: [b][n][h*64+d] f16  (input to the final projection GEMM)
// ---------------------------------------------------------------------------
__global__ void __launch_bounds__(128)
flash_attn(const _Float16* __restrict__ Qh, const _Float16* __restrict__ Kh,
           const _Float16* __restrict__ Vt, _Float16* __restrict__ Ah) {
  __shared__ _Float16 p_lds[4][16][80];  // per-wave P relayout slab

  const int bh    = blockIdx.y;
  const int b     = bh >> 4;   // / HEADS
  const int h     = bh & 15;
  const int lane  = threadIdx.x & 31;
  const int wave  = threadIdx.x >> 5;
  const int qBase = blockIdx.x * 64 + wave * 16;
  const int lrow  = lane & 15;
  const int khalf = (lane >> 4) * 8;
  const int kb2   = (lane >> 4) * 16;

  const _Float16* Qp = Qh + (size_t)bh * SEQ_ * HD_;
  const _Float16* Kp = Kh + (size_t)bh * SEQ_ * HD_;
  const _Float16* Vp = Vt + (size_t)bh * HD_ * SEQ_;

  // Q A-fragments (16 x 64 over two K=32 chunks), loaded once from global
  v16h qa[2];
#pragma unroll
  for (int c = 0; c < 2; ++c) {
    V16U u;
    const _Float16* qp = &Qp[(size_t)(qBase + lrow) * HD_ + c * 32];
    u.h[0] = *(const v8h*)&qp[khalf];
    u.h[1] = *(const v8h*)&qp[khalf + 16];
    qa[c] = u.v;
  }

  v8f acc[4] = {};        // O accumulators: 4 d-tiles of 16x16 f32
  float mrow[8], lsum[8]; // running row max / row sum (per C-layout row)
#pragma unroll
  for (int j = 0; j < 8; ++j) { mrow[j] = -1e30f; lsum[j] = 0.f; }

  for (int kb = 0; kb < SEQ_; kb += 64) {
    if (kb + 64 < SEQ_) {  // pull next K/V block toward L2/WGP$
      __builtin_prefetch(&Kp[(size_t)(kb + 64 + lane) * HD_], 0, 1);
      __builtin_prefetch(&Vp[(size_t)lane * SEQ_ + kb + 64], 0, 1);
    }

    // ---- preload ALL K and V fragments for this block (clause-able) ----
    v16h kf[4][2], vf[4][2];
#pragma unroll
    for (int n = 0; n < 4; ++n) {
      const _Float16* kp = &Kp[(size_t)(kb + n * 16 + lrow) * HD_];
      V16U u0, u1;
      u0.h[0] = *(const v8h*)&kp[khalf];
      u0.h[1] = *(const v8h*)&kp[khalf + 16];
      u1.h[0] = *(const v8h*)&kp[32 + khalf];
      u1.h[1] = *(const v8h*)&kp[32 + khalf + 16];
      kf[n][0] = u0.v; kf[n][1] = u1.v;
    }
#pragma unroll
    for (int tD = 0; tD < 4; ++tD) {
      const _Float16* vp = &Vp[(size_t)(tD * 16 + lrow) * SEQ_ + kb];
      V16U u0, u1;
      u0.h[0] = *(const v8h*)&vp[khalf];
      u0.h[1] = *(const v8h*)&vp[khalf + 16];
      u1.h[0] = *(const v8h*)&vp[32 + khalf];
      u1.h[1] = *(const v8h*)&vp[32 + khalf + 16];
      vf[tD][0] = u0.v; vf[tD][1] = u1.v;
    }

    // ---- energy tile E = Q * K^T : 16 x 64 (4 n-tiles), back-to-back ----
    v8f e[4];
#pragma unroll
    for (int n = 0; n < 4; ++n) {
      v8f z = {};
      e[n] = wmma_f16(qa[1], kf[n][1], wmma_f16(qa[0], kf[n][0], z));
    }

    // ---- online softmax: DPP row_xmask butterflies, no LDS traffic ----
#pragma unroll
    for (int j = 0; j < 8; ++j) {
      float v = e[0][j] * SCALE_;
      e[0][j] = v;
#pragma unroll
      for (int n = 1; n < 4; ++n) {
        float w = e[n][j] * SCALE_;
        e[n][j] = w;
        v = fmaxf(v, w);
      }
      v = rowmax16(v);
      float mnew  = fmaxf(mrow[j], v);
      float alpha = __expf(mrow[j] - mnew);
      mrow[j] = mnew;
      float s = 0.f;
#pragma unroll
      for (int n = 0; n < 4; ++n) {
        float p = __expf(e[n][j] - mnew);
        e[n][j] = p;
        s += p;
      }
      s = rowsum16(s);
      lsum[j] = lsum[j] * alpha + s;
#pragma unroll
      for (int tD = 0; tD < 4; ++tD) acc[tD][j] *= alpha;
    }

    // ---- P: C-layout -> LDS -> A-fragment layout ----
    const int prow = (lane >> 4) * 8;
#pragma unroll
    for (int n = 0; n < 4; ++n)
#pragma unroll
      for (int j = 0; j < 8; ++j)
        p_lds[wave][prow + j][n * 16 + lrow] = (_Float16)e[n][j];
    asm volatile("s_wait_dscnt 0" ::: "memory");  // wave-local LDS RAW fence

    v16h pa[2];
#pragma unroll
    for (int c = 0; c < 2; ++c) {
      V16U u;
      u.h[0] = *(const v8h*)&p_lds[wave][lrow][c * 32 + khalf];
      u.h[1] = *(const v8h*)&p_lds[wave][lrow][c * 32 + khalf + 16];
      pa[c] = u.v;
    }

    // ---- O += P * V (fragments already in registers) ----
#pragma unroll
    for (int tD = 0; tD < 4; ++tD)
      acc[tD] = wmma_f16(pa[1], vf[tD][1], wmma_f16(pa[0], vf[tD][0], acc[tD]));
  }

  // ---- normalize and write merged-head layout [b][n][h*64+d] ----
#pragma unroll
  for (int tD = 0; tD < 4; ++tD)
#pragma unroll
    for (int j = 0; j < 8; ++j) {
      int q = qBase + (lane >> 4) * 8 + j;
      int d = tD * 16 + lrow;
      float o = acc[tD][j] / lsum[j];
      Ah[((size_t)b * SEQ_ + q) * EMB_ + h * HD_ + d] = (_Float16)o;
    }
}

// ---------------------------------------------------------------------------
extern "C" void kernel_launch(void* const* d_in, const int* in_sizes, int n_in,
                              void* d_out, int out_size, void* d_ws, size_t ws_size,
                              hipStream_t stream) {
  (void)in_sizes; (void)n_in; (void)out_size; (void)ws_size;
  const float* x  = (const float*)d_in[0];
  const float* Wq = (const float*)d_in[1];
  const float* bq = (const float*)d_in[2];
  const float* Wk = (const float*)d_in[3];
  const float* bk = (const float*)d_in[4];
  const float* Wv = (const float*)d_in[5];
  const float* bv = (const float*)d_in[6];
  const float* Wo = (const float*)d_in[7];
  const float* bo = (const float*)d_in[8];
  float* out = (float*)d_out;

  char* ws = (char*)d_ws;
  const size_t nX = (size_t)ROWS_ * EMB_;  // 8M elements
  const size_t nW = (size_t)EMB_ * EMB_;   // 1M elements
  _Float16* xh  = (_Float16*)ws; ws += nX * 2;
  _Float16* wqh = (_Float16*)ws; ws += nW * 2;
  _Float16* wkh = (_Float16*)ws; ws += nW * 2;
  _Float16* wvh = (_Float16*)ws; ws += nW * 2;
  _Float16* woh = (_Float16*)ws; ws += nW * 2;
  _Float16* Qh  = (_Float16*)ws; ws += nX * 2;  // [b][h][n][d]
  _Float16* Kh  = (_Float16*)ws; ws += nX * 2;  // [b][h][n][d]
  _Float16* Vt  = (_Float16*)ws; ws += nX * 2;  // [b][h][d][n]
  _Float16* Ah  = (_Float16*)ws; ws += nX * 2;  // [b][n][e]

  cvt_f32_f16<<<(unsigned)(nX / 2048), 256, 0, stream>>>(x,  xh,  (int)nX);
  cvt_f32_f16<<<(unsigned)(nW / 2048), 256, 0, stream>>>(Wq, wqh, (int)nW);
  cvt_f32_f16<<<(unsigned)(nW / 2048), 256, 0, stream>>>(Wk, wkh, (int)nW);
  cvt_f32_f16<<<(unsigned)(nW / 2048), 256, 0, stream>>>(Wv, wvh, (int)nW);
  cvt_f32_f16<<<(unsigned)(nW / 2048), 256, 0, stream>>>(Wo, woh, (int)nW);

  dim3 gg(ROWS_ / 128, EMB_ / 128);
  const unsigned shmem = 4 * AS_BYTES_;  // As[2] + Bs[2] = 40960 B
  gemm128<0><<<gg, 256, shmem, stream>>>(xh, wqh, bq, (void*)Qh);
  gemm128<0><<<gg, 256, shmem, stream>>>(xh, wkh, bk, (void*)Kh);
  gemm128<2><<<gg, 256, shmem, stream>>>(xh, wvh, bv, (void*)Vt);

  flash_attn<<<dim3(SEQ_ / 64, BATCH_ * HEADS_), 128, 0, stream>>>(Qh, Kh, Vt, Ah);

  gemm128<1><<<gg, 256, shmem, stream>>>(Ah, woh, bo, (void*)out);
}
